// DecoderWithAttention_84456236909285
// MI455X (gfx1250) — matrix operand order repaired and checked by reading, not compiled, not run on hardware
//
#include <hip/hip_runtime.h>
#include <hip/hip_bf16.h>

#define NB 64      // batch
#define NP 196     // pixels (14x14)
#define NC 256     // ENC
#define NA 512     // ATT
#define ND 512     // DEC
#define NE 512     // EMBED
#define NV 30000   // VOCAB
#define NT 20      // timesteps
#define NX (NE + NC)   // 768 = LSTM input width
#define NG (4 * ND)    // 2048 = gates width
#define NTILE 4        // N-tiles per wave in the WMMA GEMM

typedef __attribute__((ext_vector_type(16))) __bf16 v16bf;
typedef __attribute__((ext_vector_type(8)))  float  v8f;

__device__ __forceinline__ __bf16 to_bf16(float f) {
  unsigned u; __builtin_memcpy(&u, &f, 4);
  u += 0x7FFFu + ((u >> 16) & 1u);           // round-to-nearest-even
  unsigned short h = (unsigned short)(u >> 16);
  __bf16 r; __builtin_memcpy(&r, &h, 2);
  return r;
}

// ---------------- WMMA fragment loaders (ISA 7.12.2, wave32) ----------------
// A 16x32 bf16: lane&15 = row M; lane half 0/1 selects K 0..15 / 16..31. Per lane this is
// one contiguous 32B chunk -> compiler emits 2x global_load_b128.
__device__ __forceinline__ v16bf load_afrag(const __bf16* __restrict__ A, int lda,
                                            int m0, int k0, int lane) {
  int row = lane & 15, half = lane >> 4;
  const __bf16* base = A + (size_t)(m0 + row) * lda + k0;
  v16bf a;
#pragma unroll
  for (int e = 0; e < 16; ++e) {
    int j = e >> 1, s = e & 1;
    int k = half * 16 + j * 2 + s;
    a[e] = base[k];
  }
  return a;
}
// B tile pre-swizzled: 512 bf16 per 32x16 tile, lane's 16 elements contiguous.
// One coalesced 1KB read per tile per wave (2x global_load_b128 per lane).
__device__ __forceinline__ v16bf load_bfrag_sw(const __bf16* __restrict__ tile,
                                               int lane) {
  const __bf16* p = tile + lane * 16;
  v16bf b;
#pragma unroll
  for (int e = 0; e < 16; ++e) b[e] = p[e];
  return b;
}

// -------- bf16 WMMA GEMM: C = A@Bsw (+ A2@B2sw) + bias (+ bias2) --------
// One wave computes a 16x64 strip (NTILE accumulators sharing each A fragment).
// Bsw layout: [Ntiles][Ktiles][512] (k-tiles contiguous per n-tile stream).
__global__ __launch_bounds__(128)
void k_wmma_gemm(const __bf16* __restrict__ A, int lda,
                 const __bf16* __restrict__ Bsw, int Ktiles,
                 const __bf16* __restrict__ A2, int lda2,
                 const __bf16* __restrict__ B2sw, int Ktiles2,
                 float* __restrict__ Cp, long long ldc,
                 const float* __restrict__ bias, const float* __restrict__ bias2,
                 int Mtiles, int Ntiles) {
  int lane = threadIdx.x & 31;
  int ngroups = (Ntiles + NTILE - 1) / NTILE;
  int wave = blockIdx.x * (blockDim.x >> 5) + (threadIdx.x >> 5);
  int mt = wave / ngroups;
  if (mt >= Mtiles) return;                 // wave-uniform: EXEC stays all-ones
  int ntb = (wave - mt * ngroups) * NTILE;
  int m0 = mt << 4;

  int nti[NTILE];
#pragma unroll
  for (int i = 0; i < NTILE; ++i) {
    int t = ntb + i;
    nti[i] = (t < Ntiles) ? t : (Ntiles - 1);   // clamp: valid addr, store guarded
  }
  v8f acc[NTILE];
#pragma unroll
  for (int i = 0; i < NTILE; ++i) acc[i] = (v8f){0.f,0.f,0.f,0.f,0.f,0.f,0.f,0.f};

  for (int kt = 0; kt < Ktiles; ++kt) {
    v16bf a = load_afrag(A, lda, m0, kt * 32, lane);
#pragma unroll
    for (int i = 0; i < NTILE; ++i) {
      const __bf16* tile = Bsw + ((size_t)nti[i] * Ktiles + kt) * 512;
      __builtin_prefetch(tile + 512 + lane * 16, 0, 1);  // next k-tile of stream
      v16bf b = load_bfrag_sw(tile, lane);
      acc[i] = __builtin_amdgcn_wmma_f32_16x16x32_bf16(false, a, false, b,
                                                       (short)0, acc[i],
                                                       false, false);
    }
  }
  if (A2) {                                  // fused second GEMM (LSTM gates)
    for (int kt = 0; kt < Ktiles2; ++kt) {
      v16bf a = load_afrag(A2, lda2, m0, kt * 32, lane);
#pragma unroll
      for (int i = 0; i < NTILE; ++i) {
        const __bf16* tile = B2sw + ((size_t)nti[i] * Ktiles2 + kt) * 512;
        v16bf b = load_bfrag_sw(tile, lane);
        acc[i] = __builtin_amdgcn_wmma_f32_16x16x32_bf16(false, a, false, b,
                                                         (short)0, acc[i],
                                                         false, false);
      }
    }
  }
  int col = lane & 15, half = lane >> 4;
#pragma unroll
  for (int i = 0; i < NTILE; ++i) {
    if (ntb + i < Ntiles) {
      int n0 = (ntb + i) << 4;
      float bv = (bias ? bias[n0 + col] : 0.f) + (bias2 ? bias2[n0 + col] : 0.f);
#pragma unroll
      for (int r = 0; r < 8; ++r)
        Cp[(size_t)(m0 + r + 8 * half) * ldc + n0 + col] = acc[i][r] + bv;
    }
  }
}

// ---------------- one-time prep kernels ----------------
__global__ __launch_bounds__(256)
void k_prep_feats(const float* __restrict__ ef, float* __restrict__ feats,
                  __bf16* __restrict__ feats_bf) {
  size_t idx = (size_t)blockIdx.x * 256 + threadIdx.x;
  if (idx >= (size_t)NB * NP * NC) return;
  int c = (int)(idx % NC);
  size_t bp = idx / NC;
  int p = (int)(bp % NP), b = (int)(bp / NP);
  float v = ef[((size_t)b * NC + c) * NP + p];   // [B,C,H,W] -> [B,P,C]
  feats[idx] = v;
  feats_bf[idx] = to_bf16(v);
}

// Swizzle an fp32 weight into per-lane WMMA B-fragment layout (bf16).
// B[k][n] = src[k*N+n] (transpose==0) or src[n*K+k] (transpose==1).
// dst layout: [Ntiles][Ktiles][512]; tile elem idx = lane*16 + e,
// k = kt*32 + (lane>>4)*16 + e,  n = nt*16 + (lane&15).
__global__ __launch_bounds__(256)
void k_swizzleB(const float* __restrict__ src, __bf16* __restrict__ dst,
                int K, int N, int transpose) {
  size_t i = (size_t)blockIdx.x * 256 + threadIdx.x;
  int Kt = K / 32, Nt2 = N / 16;
  if (i >= (size_t)Nt2 * Kt * 512) return;
  int et   = (int)(i & 511);
  size_t tile = i >> 9;
  int kt = (int)(tile % Kt), nt = (int)(tile / Kt);
  int lane = et >> 4, e = et & 15;
  int k = kt * 32 + (lane >> 4) * 16 + e;
  int n = nt * 16 + (lane & 15);
  float v = transpose ? src[(size_t)n * K + k] : src[(size_t)k * N + n];
  dst[i] = to_bf16(v);
}

__global__ __launch_bounds__(256)
void k_embed(const int* __restrict__ cap, const float* __restrict__ emb,
             __bf16* __restrict__ embs) {
  size_t i = (size_t)blockIdx.x * 256 + threadIdx.x;
  if (i >= (size_t)NB * NT * NE) return;
  int e = (int)(i % NE);
  size_t bt = i / NE;
  int tok = cap[bt];
  embs[i] = to_bf16(emb[(size_t)tok * NE + e]);
}

__global__ __launch_bounds__(256)
void k_init_state(float* h, float* c, __bf16* hbf) {
  int i = blockIdx.x * 256 + threadIdx.x;
  if (i < NB * ND) { h[i] = 0.f; c[i] = 0.f; hbf[i] = to_bf16(0.f); }
}

// ------- fused attention: relu-dot, softmax over P, weighted context -------
// One block (256 threads = 8 waves) per batch element.
__global__ __launch_bounds__(256)
void k_att(const float* __restrict__ enc_att, const float* __restrict__ dec_att,
           const float* __restrict__ W_full, const float* __restrict__ b_full,
           const float* __restrict__ feats, const __bf16* __restrict__ embs,
           __bf16* __restrict__ x_bf, int t) {
  __shared__ float s_dec[NA], s_wf[NA], s_att[NP], s_red[8];
  int b = blockIdx.x, tid = threadIdx.x, lane = tid & 31, wid = tid >> 5;
  for (int i = tid; i < NA; i += 256) {
    s_dec[i] = dec_att[(size_t)b * NA + i];
    s_wf[i]  = W_full[i];
  }
  __syncthreads();
  float bf0 = b_full[0];
  for (int p = wid; p < NP; p += 8) {
    const float* er = enc_att + ((size_t)b * NP + p) * NA;
    float sum = 0.f;
    for (int a = lane; a < NA; a += 32) {
      float v = er[a] + s_dec[a];
      sum += (v > 0.f ? v : 0.f) * s_wf[a];
    }
#pragma unroll
    for (int o = 16; o; o >>= 1) sum += __shfl_xor(sum, o, 32);
    if (lane == 0) s_att[p] = sum + bf0;
  }
  __syncthreads();
  float m = -3.4e38f;
  for (int p = tid; p < NP; p += 256) m = fmaxf(m, s_att[p]);
#pragma unroll
  for (int o = 16; o; o >>= 1) m = fmaxf(m, __shfl_xor(m, o, 32));
  if (lane == 0) s_red[wid] = m;
  __syncthreads();
  if (tid < 8) {
    float v = s_red[tid];
#pragma unroll
    for (int o = 4; o; o >>= 1) v = fmaxf(v, __shfl_xor(v, o, 32));
    if (tid == 0) s_red[0] = v;
  }
  __syncthreads();
  float gmax = s_red[0];
  __syncthreads();
  float ssum = 0.f;
  for (int p = tid; p < NP; p += 256) {
    float e = expf(s_att[p] - gmax);
    s_att[p] = e;
    ssum += e;
  }
#pragma unroll
  for (int o = 16; o; o >>= 1) ssum += __shfl_xor(ssum, o, 32);
  if (lane == 0) s_red[wid] = ssum;
  __syncthreads();
  if (tid < 8) {
    float v = s_red[tid];
#pragma unroll
    for (int o = 4; o; o >>= 1) v += __shfl_xor(v, o, 32);
    if (tid == 0) s_red[0] = v;
  }
  __syncthreads();
  float inv = 1.f / s_red[0];
  float ctx = 0.f;
  const float* fb = feats + (size_t)b * NP * NC + tid;
  for (int p = 0; p < NP; ++p) ctx += s_att[p] * fb[(size_t)p * NC];
  ctx *= inv;
  x_bf[(size_t)b * NX + NE + tid] = to_bf16(ctx);
  const __bf16* er2 = embs + ((size_t)b * NT + t) * NE;
  for (int i = tid; i < NE; i += 256) x_bf[(size_t)b * NX + i] = er2[i];
}

// ---------------- elementwise LSTM cell ----------------
__global__ __launch_bounds__(256)
void k_lstm(const float* __restrict__ gates, float* __restrict__ h,
            float* __restrict__ c, __bf16* __restrict__ hbf) {
  int idx = blockIdx.x * 256 + threadIdx.x;
  if (idx >= NB * ND) return;
  int b = idx >> 9, d = idx & (ND - 1);
  const float* g = gates + (size_t)b * NG;
  float ig = 1.f / (1.f + expf(-g[d]));
  float fg = 1.f / (1.f + expf(-g[ND + d]));
  float gg = tanhf(g[2 * ND + d]);
  float og = 1.f / (1.f + expf(-g[3 * ND + d]));
  float cn = fg * c[idx] + ig * gg;
  float hn = og * tanhf(cn);
  c[idx] = cn;
  h[idx] = hn;
  hbf[idx] = to_bf16(hn);
}

extern "C" void kernel_launch(void* const* d_in, const int* in_sizes, int n_in,
                              void* d_out, int out_size, void* d_ws, size_t ws_size,
                              hipStream_t stream) {
  const float* ef     = (const float*)d_in[0];
  const int*   cap    = (const int*)  d_in[1];
  const float* W_enc  = (const float*)d_in[2];
  const float* b_enc  = (const float*)d_in[3];
  const float* W_dec  = (const float*)d_in[4];
  const float* b_dec  = (const float*)d_in[5];
  const float* W_full = (const float*)d_in[6];
  const float* b_full = (const float*)d_in[7];
  const float* emb    = (const float*)d_in[8];
  const float* W_ih   = (const float*)d_in[9];
  const float* b_ih   = (const float*)d_in[10];
  const float* W_hh   = (const float*)d_in[11];
  const float* b_hh   = (const float*)d_in[12];
  const float* W_fc   = (const float*)d_in[13];
  const float* b_fc   = (const float*)d_in[14];
  float* out = (float*)d_out;
  char*  ws  = (char*)d_ws;

  size_t o = 0;
  auto alloc = [&](size_t bytes) {
    size_t r = o; o += (bytes + 255) & ~(size_t)255; return r;
  };
  float*  feats    = (float*) (ws + alloc((size_t)NB * NP * NC * 4));
  __bf16* feats_bf = (__bf16*)(ws + alloc((size_t)NB * NP * NC * 2));
  float*  enc_att  = (float*) (ws + alloc((size_t)NB * NP * NA * 4));
  __bf16* Wenc_sw  = (__bf16*)(ws + alloc((size_t)NC * NA * 2));
  __bf16* Wdec_sw  = (__bf16*)(ws + alloc((size_t)ND * NA * 2));
  __bf16* Wih_sw   = (__bf16*)(ws + alloc((size_t)NX * NG * 2));
  __bf16* Whh_sw   = (__bf16*)(ws + alloc((size_t)ND * NG * 2));
  __bf16* Wfc_sw   = (__bf16*)(ws + alloc((size_t)ND * NV * 2));
  __bf16* embs     = (__bf16*)(ws + alloc((size_t)NB * NT * NE * 2));
  float*  h        = (float*) (ws + alloc((size_t)NB * ND * 4));
  float*  c        = (float*) (ws + alloc((size_t)NB * ND * 4));
  __bf16* hbf      = (__bf16*)(ws + alloc((size_t)NB * ND * 2));
  float*  dec_att  = (float*) (ws + alloc((size_t)NB * NA * 4));
  __bf16* x_bf     = (__bf16*)(ws + alloc((size_t)NB * NX * 2));
  float*  gates    = (float*) (ws + alloc((size_t)NB * NG * 4));
  (void)ws_size; (void)in_sizes; (void)n_in; (void)out_size;

  auto nblk = [](size_t n, int tpb) { return (int)((n + (size_t)tpb - 1) / tpb); };
  auto gemm_blocks = [](int Mtiles, int Ntiles) {
    int waves = Mtiles * ((Ntiles + NTILE - 1) / NTILE);
    return (waves * 32 + 127) / 128;
  };

  // ---- one-time prep ----
  k_prep_feats<<<nblk((size_t)NB * NP * NC, 256), 256, 0, stream>>>(ef, feats, feats_bf);
  k_swizzleB<<<nblk((size_t)NC * NA, 256), 256, 0, stream>>>(W_enc, Wenc_sw, NC, NA, 0);
  k_swizzleB<<<nblk((size_t)ND * NA, 256), 256, 0, stream>>>(W_dec, Wdec_sw, ND, NA, 0);
  k_swizzleB<<<nblk((size_t)ND * NV, 256), 256, 0, stream>>>(W_fc, Wfc_sw, ND, NV, 0);
  k_swizzleB<<<nblk((size_t)NX * NG, 256), 256, 0, stream>>>(W_ih, Wih_sw, NX, NG, 1);
  k_swizzleB<<<nblk((size_t)ND * NG, 256), 256, 0, stream>>>(W_hh, Whh_sw, ND, NG, 1);
  k_embed<<<nblk((size_t)NB * NT * NE, 256), 256, 0, stream>>>(cap, emb, embs);
  k_init_state<<<nblk(NB * ND, 256), 256, 0, stream>>>(h, c, hbf);

  // enc_att[B*P, A] = feats_bf[B*P, C] @ W_enc + b_enc
  {
    int Mt = (NB * NP) / 16, Nt = NA / 16;                 // 784 x 32 tiles
    k_wmma_gemm<<<gemm_blocks(Mt, Nt), 128, 0, stream>>>(
        feats_bf, NC, Wenc_sw, NC / 32,
        nullptr, 0, nullptr, 0,
        enc_att, (long long)NA, b_enc, nullptr, Mt, Nt);
  }

  // ---- sequential decode: T steps ----
  for (int t = 0; t < NT; ++t) {
    // dec_att[B, A] = hbf @ W_dec + b_dec
    k_wmma_gemm<<<gemm_blocks(4, NA / 16), 128, 0, stream>>>(
        hbf, ND, Wdec_sw, ND / 32,
        nullptr, 0, nullptr, 0,
        dec_att, (long long)NA, b_dec, nullptr, 4, NA / 16);
    // attention + softmax + context + assemble x = [emb | ctx]
    k_att<<<NB, 256, 0, stream>>>(enc_att, dec_att, W_full, b_full,
                                  feats, embs, x_bf, t);
    // gates[B, 4D] = x @ W_ih^T + h @ W_hh^T + b_ih + b_hh
    k_wmma_gemm<<<gemm_blocks(4, NG / 16), 128, 0, stream>>>(
        x_bf, NX, Wih_sw, NX / 32,
        hbf, ND, Whh_sw, ND / 32,
        gates, (long long)NG, b_ih, b_hh, 4, NG / 16);
    // LSTM cell update
    k_lstm<<<nblk(NB * ND, 256), 256, 0, stream>>>(gates, h, c, hbf);
    // out[b, t, :] = h_new @ W_fc + b_fc  (ldc = T*V indexes [B][T][V])
    k_wmma_gemm<<<gemm_blocks(4, NV / 16), 128, 0, stream>>>(
        hbf, ND, Wfc_sw, ND / 32,
        nullptr, 0, nullptr, 0,
        out + (size_t)t * NV, (long long)NT * NV, b_fc, nullptr, 4, NV / 16);
  }
}
